// LSTMDecoder_69028714381825
// MI455X (gfx1250) — compile-verified, matrix operand order checked by reading, not run on hardware
//
#include <hip/hip_runtime.h>
#include <math.h>

// Problem constants (match reference)
#define VV 10000
#define EE 512
#define HH 1024
#define BB 64
#define TT 128
#define SS 64

typedef __attribute__((ext_vector_type(16))) __bf16 v16bf;
typedef __attribute__((ext_vector_type(8)))  float  v8f;

__device__ __forceinline__ unsigned short f2bfu(float f) {
    return __builtin_bit_cast(unsigned short, (__bf16)f);
}

// ---------------------------------------------------------------------------
// WMMA GEMM: C[M x N] = act( A1*W1^T + A2*W2^T + bias (+C if accumulate) )
//   A: bf16 activations (row-major M x K)
//   W: bf16 weights (row-major N x K, pre-converted once into workspace)
// MB = register blocking in M: one wave computes a (16*MB) x 16 strip,
// reusing each B fragment across MB accumulators (MB wmmas per k-step).
// Fragment packing per CDNA5 ISA 7.12.2 (16-bit A 16x32 / B 32x16 layouts):
// lanes 0-15 hold K {0..7,16..23}, lanes 16-31 hold K {8..15,24..31};
// each half is one contiguous 16-byte chunk -> two global_load_b128 per frag.
// ---------------------------------------------------------------------------
template <int MB>
__device__ __forceinline__ void wmma_accum_src_bf(const unsigned short* __restrict__ A, long lda,
                                                  const unsigned short* __restrict__ W, long ldw,
                                                  int K, int arow0, int wrow, int koff,
                                                  v8f acc[MB]) {
    const unsigned short* __restrict__ Wr = W + (long)wrow * ldw;
    for (int kk = 0; kk < K; kk += 32) {
        union { v16bf bf; uint4 q[2]; } b;
        b.q[0] = *(const uint4*)(Wr + kk + koff);
        b.q[1] = *(const uint4*)(Wr + kk + 16 + koff);
#pragma unroll
        for (int m = 0; m < MB; ++m) {
            const unsigned short* __restrict__ Ar = A + (long)(arow0 + 16 * m) * lda;
            union { v16bf bf; uint4 q[2]; } a;
            a.q[0] = *(const uint4*)(Ar + kk + koff);
            a.q[1] = *(const uint4*)(Ar + kk + 16 + koff);
            acc[m] = __builtin_amdgcn_wmma_f32_16x16x32_bf16(
                false, a.bf, false, b.bf, (short)0, acc[m], false, false);
        }
    }
}

template <int MB>
__global__ void wmma_gemm_dual(const unsigned short* A1, long lda1,
                               const unsigned short* W1, long ldw1, int K1,
                               const unsigned short* A2, long lda2,
                               const unsigned short* W2, long ldw2, int K2,
                               const float* bias, float* C, unsigned short* Cbf, long ldc,
                               int M, int N, int accumulate, int act) {
    int wave = threadIdx.x >> 5;
    int lane = threadIdx.x & 31;
    int tile = blockIdx.x * (blockDim.x >> 5) + wave;
    int ntn  = N >> 4;
    int ntmg = M / (16 * MB);
    if (tile >= ntmg * ntn) return;         // wave-uniform: EXEC stays all-ones
    int mbase = (tile / ntn) * 16 * MB;
    int nbase = (tile % ntn) << 4;
    int l15  = lane & 15;
    int hi   = (lane >= 16) ? 8 : 0;        // K-half select for A/B fragments
    int arow0 = mbase + l15;
    int wrow  = nbase + l15;

    v8f acc[MB];
#pragma unroll
    for (int m = 0; m < MB; ++m) acc[m] = {};

    if (A1) wmma_accum_src_bf<MB>(A1, lda1, W1, ldw1, K1, arow0, wrow, hi, acc);
    if (A2) wmma_accum_src_bf<MB>(A2, lda2, W2, ldw2, K2, arow0, wrow, hi, acc);

    float bv = bias ? bias[wrow] : 0.0f;
#pragma unroll
    for (int m = 0; m < MB; ++m) {
#pragma unroll
        for (int v = 0; v < 8; ++v) {
            int r = mbase + 16 * m + v + hi; // C/D layout: VGPR v -> row v (+8 hi lanes)
            long off = (long)r * ldc + nbase + l15;
            float val = acc[m][v] + bv;
            if (accumulate) val += C[off];
            if (act) val = tanhf(val);
            if (C)   C[off]   = val;
            if (Cbf) Cbf[off] = f2bfu(val);
        }
    }
}

// ---------------------------------------------------------------------------
// Small helper kernels
// ---------------------------------------------------------------------------
__global__ void f32_to_bf16(const float* __restrict__ s, unsigned short* __restrict__ d, long n) {
    long i = (long)blockIdx.x * blockDim.x + threadIdx.x;
    if (i < n) d[i] = f2bfu(s[i]);
}

__global__ void embed_gather_bf(const int* __restrict__ tok, const float* __restrict__ emb,
                                unsigned short* __restrict__ x, int BT, int E) {
    long i = (long)blockIdx.x * blockDim.x + threadIdx.x;
    if (i >= (long)BT * E) return;
    long bt = i / E;
    int  e  = (int)(i - bt * E);
    x[i] = f2bfu(emb[(long)tok[bt] * E + e]);
}

__global__ void bias_sum2(const float* a0, const float* b0, float* o0,
                          const float* a1, const float* b1, float* o1, int n) {
    int i = blockIdx.x * blockDim.x + threadIdx.x;
    if (i < n) { o0[i] = a0[i] + b0[i]; o1[i] = a1[i] + b1[i]; }
}

// gates: (B, 4H) with PyTorch order i,f,g,o ; c fp32, h emitted bf16
__global__ void lstm_act(const float* __restrict__ gates, const float* __restrict__ c_prev,
                         float* __restrict__ c_new, unsigned short* __restrict__ h_new) {
    int i = blockIdx.x * blockDim.x + threadIdx.x;
    if (i >= BB * HH) return;
    int b = i >> 10;            // /H
    int h = i & (HH - 1);
    const float* g = gates + (long)b * 4 * HH;
    float ig = g[h], fg = g[HH + h], gg = g[2 * HH + h], og = g[3 * HH + h];
    float si = 1.0f / (1.0f + expf(-ig));
    float sf = 1.0f / (1.0f + expf(-fg));
    float so = 1.0f / (1.0f + expf(-og));
    float c2 = sf * c_prev[i] + si * tanhf(gg);
    c_new[i] = c2;
    h_new[i] = f2bfu(so * tanhf(c2));
}

// One block per batch element; 64 threads (= S). scores -> softmax(S) -> ctx(bf16).
__global__ void attn_kernel(const float* __restrict__ enc, const unsigned char* __restrict__ mask,
                            const float* __restrict__ q, unsigned short* __restrict__ ctx) {
    int b = blockIdx.x;
    int s = threadIdx.x;
    __shared__ float sraw[SS];
    __shared__ float sexp[SS];

    const float* er = enc + ((long)s * BB + b) * HH;
    const float* qr = q + (long)b * HH;
    float sc = 0.0f;
    for (int h = 0; h < HH; ++h) sc += er[h] * qr[h];
    if (mask[s * BB + b]) sc = -INFINITY;
    sraw[s] = sc;
    __syncthreads();

    float mx = -INFINITY;
    for (int t = 0; t < SS; ++t) mx = fmaxf(mx, sraw[t]);
    sexp[s] = expf(sraw[s] - mx);
    __syncthreads();

    float sum = 0.0f;
    for (int t = 0; t < SS; ++t) sum += sexp[t];
    float inv = 1.0f / sum;

    for (int h = s; h < HH; h += SS) {
        float acc = 0.0f;
        for (int t = 0; t < SS; ++t)
            acc += sexp[t] * enc[((long)t * BB + b) * HH + h];
        ctx[(long)b * HH + h] = f2bfu(acc * inv);
    }
}

// ---------------------------------------------------------------------------
// Orchestration
// ---------------------------------------------------------------------------
extern "C" void kernel_launch(void* const* d_in, const int* in_sizes, int n_in,
                              void* d_out, int out_size, void* d_ws, size_t ws_size,
                              hipStream_t stream) {
    const int*           tok      = (const int*)d_in[0];
    const float*         enc_outs = (const float*)d_in[1];
    const float*         enc_h    = (const float*)d_in[2];
    const float*         enc_c    = (const float*)d_in[3];
    const unsigned char* pad      = (const unsigned char*)d_in[4];
    const float*         emb      = (const float*)d_in[5];
    const float* W_ih0 = (const float*)d_in[6];
    const float* W_hh0 = (const float*)d_in[7];
    const float* b_ih0 = (const float*)d_in[8];
    const float* b_hh0 = (const float*)d_in[9];
    const float* W_ih1 = (const float*)d_in[10];
    const float* W_hh1 = (const float*)d_in[11];
    const float* b_ih1 = (const float*)d_in[12];
    const float* b_hh1 = (const float*)d_in[13];
    const float* W_ain = (const float*)d_in[14];
    const float* W_aout= (const float*)d_in[15];
    const float* W_fc  = (const float*)d_in[16];
    const float* b_fc  = (const float*)d_in[17];
    float* logits = (float*)d_out;

    const long BH = (long)BB * HH;

    // ---- Workspace layout (byte-based, 256B-aligned chunks) ---------------
    char* base = (char*)d_ws;
    size_t ob = 0;
    auto alloc = [&](size_t bytes) -> void* {
        void* p = base + ob;
        ob = (ob + bytes + 255) & ~(size_t)255;
        return p;
    };
    unsigned short* xbf   = (unsigned short*)alloc((size_t)BB * TT * EE * 2); // (B,T,E) bf16
    float*          gates = (float*)alloc((size_t)BB * 4 * HH * 4);           // (B,4H) fp32
    unsigned short* h0buf = (unsigned short*)alloc(2 * BH * 2);               // ping-pong bf16
    unsigned short* h1buf = (unsigned short*)alloc(2 * BH * 2);
    float*          c0buf = (float*)alloc(2 * BH * 4);                        // fp32
    float*          c1buf = (float*)alloc(2 * BH * 4);
    float*          qbuf  = (float*)alloc(BH * 4);
    unsigned short* ctxb  = (unsigned short*)alloc(BH * 2);
    unsigned short* feats = (unsigned short*)alloc((size_t)BB * TT * HH * 2); // (B,T,H) bf16
    float*          bsum0 = (float*)alloc(4 * HH * 4);
    float*          bsum1 = (float*)alloc(4 * HH * 4);

    const long nW_ih0 = 4L * HH * (EE + HH);
    const long nW_hh0 = 4L * HH * HH;
    const long nW_ih1 = 4L * HH * HH;
    const long nW_hh1 = 4L * HH * HH;
    const long nW_ain = (long)HH * HH;
    const long nW_aout= (long)HH * 2 * HH;
    const long nW_fc  = (long)VV * HH;
    unsigned short* bW_ih0 = (unsigned short*)alloc(nW_ih0 * 2);
    unsigned short* bW_hh0 = (unsigned short*)alloc(nW_hh0 * 2);
    unsigned short* bW_ih1 = (unsigned short*)alloc(nW_ih1 * 2);
    unsigned short* bW_hh1 = (unsigned short*)alloc(nW_hh1 * 2);
    unsigned short* bW_ain = (unsigned short*)alloc(nW_ain * 2);
    unsigned short* bW_aout= (unsigned short*)alloc(nW_aout * 2);
    unsigned short* bW_fc  = (unsigned short*)alloc(nW_fc * 2);

    // ---- One-time prep ----------------------------------------------------
    hipMemcpyAsync(c0buf, enc_c,      BH * sizeof(float), hipMemcpyDeviceToDevice, stream);
    hipMemcpyAsync(c1buf, enc_c + BH, BH * sizeof(float), hipMemcpyDeviceToDevice, stream);

#define CVT(src, dst, n) f32_to_bf16<<<(int)(((n) + 255) / 256), 256, 0, stream>>>(src, dst, n)
    CVT(enc_h,      h0buf, BH);      // initial h0 (into ping slot 0)
    CVT(enc_h + BH, h1buf, BH);      // initial h1
    CVT(W_ih0, bW_ih0, nW_ih0);
    CVT(W_hh0, bW_hh0, nW_hh0);
    CVT(W_ih1, bW_ih1, nW_ih1);
    CVT(W_hh1, bW_hh1, nW_hh1);
    CVT(W_ain, bW_ain, nW_ain);
    CVT(W_aout, bW_aout, nW_aout);
    CVT(W_fc,  bW_fc,  nW_fc);
#undef CVT

    embed_gather_bf<<<(BB * TT * EE + 255) / 256, 256, 0, stream>>>(tok, emb, xbf, BB * TT, EE);
    bias_sum2<<<(4 * HH + 255) / 256, 256, 0, stream>>>(b_ih0, b_hh0, bsum0,
                                                        b_ih1, b_hh1, bsum1, 4 * HH);

    // GEMM grids: 4 waves per 128-thread block, one 16x16 tile per wave (MB=1)
    const int gGates = (4 * (4 * HH / 16)) / 4;   // M=64,N=4096 -> 1024 tiles -> 256 blocks
    const int gProj  = (4 * (HH / 16)) / 4;       // M=64,N=1024 -> 256 tiles -> 64 blocks
    const int gAct   = (BB * HH + 255) / 256;
    unsigned short* const NOBF = nullptr;

    // ---- Recurrence -------------------------------------------------------
    for (int t = 0; t < TT; ++t) {
        int rp = t & 1, wp = (t + 1) & 1;
        const unsigned short* feed = (t == 0) ? nullptr : (feats + (long)(t - 1) * HH); // lda=T*H

        // LSTM cell 0: gates = [x_t, feed] @ W_ih0^T + h0 @ W_hh0^T + (b_ih0+b_hh0)
        wmma_gemm_dual<1><<<gGates, 128, 0, stream>>>(
            xbf + (long)t * EE, (long)TT * EE, bW_ih0,      (long)(EE + HH), EE,
            feed,               (long)TT * HH, bW_ih0 + EE, (long)(EE + HH), HH,
            bsum0, gates, NOBF, 4L * HH, BB, 4 * HH, /*acc=*/0, /*act=*/0);
        wmma_gemm_dual<1><<<gGates, 128, 0, stream>>>(
            h0buf + rp * BH, (long)HH, bW_hh0, (long)HH, HH,
            nullptr, 0, nullptr, 0, 0,
            nullptr, gates, NOBF, 4L * HH, BB, 4 * HH, /*acc=*/1, /*act=*/0);
        lstm_act<<<gAct, 256, 0, stream>>>(gates, c0buf + rp * BH,
                                           c0buf + wp * BH, h0buf + wp * BH);

        // LSTM cell 1: gates = h0_new @ W_ih1^T + h1 @ W_hh1^T + (b_ih1+b_hh1)
        wmma_gemm_dual<1><<<gGates, 128, 0, stream>>>(
            h0buf + wp * BH, (long)HH, bW_ih1, (long)HH, HH,
            h1buf + rp * BH, (long)HH, bW_hh1, (long)HH, HH,
            bsum1, gates, NOBF, 4L * HH, BB, 4 * HH, 0, 0);
        lstm_act<<<gAct, 256, 0, stream>>>(gates, c1buf + rp * BH,
                                           c1buf + wp * BH, h1buf + wp * BH);
        unsigned short* h1n = h1buf + wp * BH;

        // Attention: q = h1 @ W_attn_in^T ; softmax over S ; ctx (bf16)
        wmma_gemm_dual<1><<<gProj, 128, 0, stream>>>(
            h1n, (long)HH, bW_ain, (long)HH, HH,
            nullptr, 0, nullptr, 0, 0,
            nullptr, qbuf, NOBF, (long)HH, BB, HH, 0, 0);
        attn_kernel<<<BB, SS, 0, stream>>>(enc_outs, pad, qbuf, ctxb);

        // out = tanh([ctx; h1] @ W_attn_out^T) -> feats[:, t, :] bf16 (next feed)
        wmma_gemm_dual<1><<<gProj, 128, 0, stream>>>(
            ctxb, (long)HH, bW_aout,      2L * HH, HH,
            h1n,  (long)HH, bW_aout + HH, 2L * HH, HH,
            nullptr, nullptr, feats + (long)t * HH, (long)TT * HH, BB, HH, 0, /*tanh=*/1);
    }

    // ---- logits = feats @ W_fc^T + b_fc : M=8192, N=10000 (=625*16), K=1024
    // MB=4: one wave computes a 64x16 strip, B fragment reused across 4 wmmas.
    const int tilesFC = (BB * TT / 64) * (VV / 16);   // 128 * 625 = 80000 waves
    wmma_gemm_dual<4><<<tilesFC / 4, 128, 0, stream>>>(
        feats, (long)HH, bW_fc, (long)HH, HH,
        nullptr, 0, nullptr, 0, 0,
        b_fc, logits, NOBF, (long)VV, BB * TT, VV, 0, 0);
}
